// GCN_88132728914210
// MI455X (gfx1250) — matrix operand order repaired and checked by reading, not compiled, not run on hardware
//
#include <hip/hip_runtime.h>
#include <hip/hip_bf16.h>

typedef __attribute__((ext_vector_type(16))) __bf16 v16bf;
typedef __attribute__((ext_vector_type(8)))  float  v8f;

#define F_IN  128
#define F_HID 64
#define F_OUT 2

// ---------------- degree / normalization ----------------

__global__ void k_init_deg(float* deg, int n) {
    int i = blockIdx.x * blockDim.x + threadIdx.x;
    if (i < n) deg[i] = 1.0f;               // self-loop contributes 1
}

__global__ void k_degree(const long long* __restrict__ ei, int E, float* deg) {
    int e = blockIdx.x * blockDim.x + threadIdx.x;
    if (e < E) {
        int d = (int)ei[(size_t)E + e];     // dst row
        unsafeAtomicAdd(&deg[d], 1.0f);     // global_atomic_add_f32 (no-return)
    }
}

__global__ void k_dinv(float* deg, int n) {
    int i = blockIdx.x * blockDim.x + threadIdx.x;
    if (i < n) deg[i] = rsqrtf(deg[i]);     // deg >= 1 always
}

// ---------------- pack W1 into WMMA B-fragment layout (bf16) ----------------
// Chunk c = kb*4 + nb covers K rows [kb*32, kb*32+32) and N cols [nb*16, nb*16+16).
// Lane l (0..15): col n = l,     K = kb*32 + {0..7} U {16..23}   (e = 0..15)
// Lane l (16..31): col n = l-16, K = kb*32 + {8..15} U {24..31}
// Stored lane-major: w1p[(c*32 + l)*16 + e]

__global__ void k_pack_w1(const float* __restrict__ W1, __bf16* __restrict__ w1p) {
    int t = blockIdx.x * blockDim.x + threadIdx.x;
    if (t >= 16 * 32) return;
    int c  = t >> 5;
    int l  = t & 31;
    int kb = c >> 2, nb = c & 3;
    int n     = nb * 16 + (l & 15);
    int kbase = kb * 32 + ((l < 16) ? 0 : 8);
    __bf16* o = w1p + (size_t)t * 16;
#pragma unroll
    for (int e = 0; e < 16; ++e)
        o[e] = (__bf16)W1[(kbase + (e < 8 ? e : e + 8)) * F_HID + n];
}

// ---------------- GEMM1: h[n,64] = x[n,128] @ W1  via v_wmma_f32_16x16x32_bf16 ----
// One wave per 16-row tile. A-fragment per ISA 16-bit A layout:
// lanes 0..15  : row m=l,    K = {k0.. k0+7} U {k0+16..k0+23}, k0 = kb*32
// lanes 16..31 : row m=l-16, K = {k0+8..k0+15} U {k0+24..k0+31}

__global__ void __launch_bounds__(32)
k_gemm1(const float* __restrict__ x, const __bf16* __restrict__ w1p,
        float* __restrict__ h, int n) {
    const int lane = threadIdx.x;
    const int hi   = (lane >= 16);
    int m = blockIdx.x * 16 + (lane & 15);
    if (m >= n) m = n - 1;                       // clamp loads (n % 16 == 0 anyway)
    const float* __restrict__ xrow = x + (size_t)m * F_IN;
    const v16bf* __restrict__ bp   = (const v16bf*)w1p;

    v8f acc0 = {}, acc1 = {}, acc2 = {}, acc3 = {};

#pragma unroll
    for (int kb = 0; kb < 4; ++kb) {
        const int k0 = kb * 32 + (hi ? 8 : 0);
        v16bf a;
#pragma unroll
        for (int e = 0; e < 8; ++e) {
            a[e]     = (__bf16)xrow[k0 + e];
            a[e + 8] = (__bf16)xrow[k0 + 16 + e];
        }
        v16bf b0 = bp[(kb * 4 + 0) * 32 + lane];
        v16bf b1 = bp[(kb * 4 + 1) * 32 + lane];
        v16bf b2 = bp[(kb * 4 + 2) * 32 + lane];
        v16bf b3 = bp[(kb * 4 + 3) * 32 + lane];
        acc0 = __builtin_amdgcn_wmma_f32_16x16x32_bf16(false, a, false, b0, (short)0, acc0, false, false);
        acc1 = __builtin_amdgcn_wmma_f32_16x16x32_bf16(false, a, false, b1, (short)0, acc1, false, false);
        acc2 = __builtin_amdgcn_wmma_f32_16x16x32_bf16(false, a, false, b2, (short)0, acc2, false, false);
        acc3 = __builtin_amdgcn_wmma_f32_16x16x32_bf16(false, a, false, b3, (short)0, acc3, false, false);
    }

    // C/D layout: VGPR r, lane l -> (M = r + (l<16?0:8), N = l&15)
    const int mbase = blockIdx.x * 16 + (hi ? 8 : 0);
    const int ncol  = lane & 15;
#pragma unroll
    for (int r = 0; r < 8; ++r) {
        int row = mbase + r;
        if (row < n) {
            float* hr = h + (size_t)row * F_HID + ncol;
            hr[0]  = acc0[r];
            hr[16] = acc1[r];
            hr[32] = acc2[r];
            hr[48] = acc3[r];
        }
    }
}

// ---------------- aggregation, layer 1 ----------------

__global__ void k_selfloop64(const float* __restrict__ h, const float* __restrict__ dinv,
                             float* __restrict__ out1, int n) {
    int i = blockIdx.x * blockDim.x + threadIdx.x;
    if (i < n * F_HID) {
        float di = dinv[i >> 6];
        out1[i] = h[i] * di * di;
    }
}

__global__ void k_edge_agg64(const long long* __restrict__ ei, int E,
                             const float* __restrict__ h, const float* __restrict__ dinv,
                             float* __restrict__ out1) {
    long long t = (long long)blockIdx.x * blockDim.x + threadIdx.x;
    long long e = t >> 4;                 // 16 threads per edge, 4 floats each
    if (e >= E) return;
    int sub = (int)(t & 15);
    int s = (int)ei[e];
    int d = (int)ei[(size_t)E + e];
    float norm = dinv[s] * dinv[d];
    const float4 v = *(const float4*)(h + (size_t)s * F_HID + sub * 4);
    float* o = out1 + (size_t)d * F_HID + sub * 4;
    unsafeAtomicAdd(o + 0, v.x * norm);
    unsafeAtomicAdd(o + 1, v.y * norm);
    unsafeAtomicAdd(o + 2, v.z * norm);
    unsafeAtomicAdd(o + 3, v.w * norm);
}

__global__ void k_bias_relu(float* out1, const float* __restrict__ b1, int n) {
    int i = blockIdx.x * blockDim.x + threadIdx.x;
    if (i < n * F_HID) {
        float v = out1[i] + b1[i & (F_HID - 1)];
        out1[i] = v > 0.0f ? v : 0.0f;
    }
}

// ---------------- layer 2 (64 -> 2): plain FMA, N too small for WMMA ----------------

__global__ void k_gemm2(const float* __restrict__ hr, const float* __restrict__ W2,
                        float* __restrict__ z, int n) {
    int i = blockIdx.x * blockDim.x + threadIdx.x;
    if (i >= n) return;
    const float* row = hr + (size_t)i * F_HID;
    float a0 = 0.0f, a1 = 0.0f;
#pragma unroll
    for (int j = 0; j < F_HID; ++j) {
        float v = row[j];
        a0 = fmaf(v, W2[2 * j + 0], a0);
        a1 = fmaf(v, W2[2 * j + 1], a1);
    }
    z[2 * i + 0] = a0;
    z[2 * i + 1] = a1;
}

__global__ void k_out_init(const float* __restrict__ z, const float* __restrict__ dinv,
                           const float* __restrict__ b2, float* __restrict__ out, int n) {
    int i = blockIdx.x * blockDim.x + threadIdx.x;
    if (i < n * F_OUT) {
        float di = dinv[i >> 1];
        out[i] = z[i] * di * di + b2[i & 1];
    }
}

__global__ void k_edge_agg2(const long long* __restrict__ ei, int E,
                            const float* __restrict__ z, const float* __restrict__ dinv,
                            float* __restrict__ out) {
    long long t = (long long)blockIdx.x * blockDim.x + threadIdx.x;
    long long e = t >> 1;
    if (e >= E) return;
    int c = (int)(t & 1);
    int s = (int)ei[e];
    int d = (int)ei[(size_t)E + e];
    float norm = dinv[s] * dinv[d];
    unsafeAtomicAdd(&out[(size_t)d * F_OUT + c], z[(size_t)s * F_OUT + c] * norm);
}

// ---------------- launcher ----------------

static inline size_t alup(size_t x) { return (x + 255) & ~(size_t)255; }

extern "C" void kernel_launch(void* const* d_in, const int* in_sizes, int n_in,
                              void* d_out, int out_size, void* d_ws, size_t ws_size,
                              hipStream_t stream) {
    const float*     x   = (const float*)d_in[0];
    const long long* ei  = (const long long*)d_in[1];   // int64 edge_index [2, E]
    const float*     W1  = (const float*)d_in[2];
    const float*     b1  = (const float*)d_in[3];
    const float*     W2  = (const float*)d_in[4];
    const float*     b2  = (const float*)d_in[5];
    float*           out = (float*)d_out;

    const int n = in_sizes[0] / F_IN;     // 100000
    const int E = in_sizes[1] / 2;        // 3200000

    // workspace carve-up
    char* w = (char*)d_ws;
    float*  deg  = (float*)w;  w += alup((size_t)n * sizeof(float));
    __bf16* w1p  = (__bf16*)w; w += alup((size_t)16 * 32 * 16 * sizeof(__bf16));
    float*  h    = (float*)w;  w += alup((size_t)n * F_HID * sizeof(float));
    float*  out1 = (float*)w;  w += alup((size_t)n * F_HID * sizeof(float));
    float*  z    = (float*)w;  w += alup((size_t)n * F_OUT * sizeof(float));
    (void)ws_size; (void)n_in; (void)out_size;

    const int T = 256;
    // normalization
    k_init_deg<<<(n + T - 1) / T, T, 0, stream>>>(deg, n);
    k_degree  <<<(E + T - 1) / T, T, 0, stream>>>(ei, E, deg);
    k_dinv    <<<(n + T - 1) / T, T, 0, stream>>>(deg, n);

    // layer 1: WMMA GEMM + scatter
    k_pack_w1 <<<2, T, 0, stream>>>(W1, w1p);
    k_gemm1   <<<(n + 15) / 16, 32, 0, stream>>>(x, w1p, h, n);
    {
        int tot = n * F_HID;
        k_selfloop64<<<(tot + T - 1) / T, T, 0, stream>>>(h, deg, out1, n);
        long long et = (long long)E * 16;
        k_edge_agg64<<<(unsigned)((et + T - 1) / T), T, 0, stream>>>(ei, E, h, deg, out1);
        k_bias_relu <<<(tot + T - 1) / T, T, 0, stream>>>(out1, b1, n);
    }

    // layer 2: small GEMM + scatter into d_out
    k_gemm2<<<(n + T - 1) / T, T, 0, stream>>>(out1, W2, z, n);
    k_out_init<<<(n * F_OUT + T - 1) / T, T, 0, stream>>>(z, deg, b2, out, n);
    {
        long long et = (long long)E * 2;
        k_edge_agg2<<<(unsigned)((et + T - 1) / T), T, 0, stream>>>(ei, E, z, deg, out);
    }
}
